// TokenChoiceTopKRouter_28028956574146
// MI455X (gfx1250) — matrix operand order repaired and checked by reading, not compiled, not run on hardware
//
#include <hip/hip_runtime.h>
#include <hip/hip_bf16.h>

typedef __attribute__((ext_vector_type(16))) _Float16 v16h;
typedef __attribute__((ext_vector_type(8)))  float    v8f;

#define D_DIM     2048
#define E_EXPERTS 256
#define TOPK      8
#define NGROUPS   8
#define GSIZE     32
#define NLIM      4
#define RSCALE    2.5f
#define NEG_BIG   (-1.0e30f)
#define NT        32          // tokens per block (two 16-row M tiles)

__global__ void zero_hist_kernel(float* __restrict__ hist) {
    int t = threadIdx.x;
    if (t < E_EXPERTS) hist[t] = 0.0f;
}

__device__ __forceinline__ v16h cvt16(float4 q0, float4 q1, float4 q2, float4 q3) {
    v16h v;
    v[0]  = (_Float16)q0.x; v[1]  = (_Float16)q0.y;
    v[2]  = (_Float16)q0.z; v[3]  = (_Float16)q0.w;
    v[4]  = (_Float16)q1.x; v[5]  = (_Float16)q1.y;
    v[6]  = (_Float16)q1.z; v[7]  = (_Float16)q1.w;
    v[8]  = (_Float16)q2.x; v[9]  = (_Float16)q2.y;
    v[10] = (_Float16)q2.z; v[11] = (_Float16)q2.w;
    v[12] = (_Float16)q3.x; v[13] = (_Float16)q3.y;
    v[14] = (_Float16)q3.z; v[15] = (_Float16)q3.w;
    return v;
}

// A-matrix 16x32 f16 layout: lane(m,hf) needs K = hf*8+0..7 and 16+hf*8+0..7
// p must already point at  row_base + k0 + hf*8
__device__ __forceinline__ v16h load_a(const float* p) {
    return cvt16(*(const float4*)(p),      *(const float4*)(p + 4),
                 *(const float4*)(p + 16), *(const float4*)(p + 20));
}

// B-matrix 32x16 f16 layout: lane(n,hf) needs K = hf*8+0..7 and 16+hf*8+0..7
// p must already point at  expert_row + k0 + hf*8
__device__ __forceinline__ v16h load_b(const float* p) {
    return cvt16(*(const float4*)(p),     *(const float4*)(p + 4),
                 *(const float4*)(p + 16), *(const float4*)(p + 20));
}

__device__ __forceinline__ float sigmoidf(float v) {
    return 1.0f / (1.0f + __expf(-v));
}

__global__ __launch_bounds__(256) void router_fused_kernel(
    const float* __restrict__ x,      // (T, 2048)
    const float* __restrict__ W,      // (256, 2048)
    const float* __restrict__ bias,   // (256,)
    float* __restrict__ out_scores,   // (T, 8)
    float* __restrict__ out_sel,      // (T, 8) indices as float
    float* __restrict__ hist)         // (256,)
{
    __shared__ float s_scores[NT * E_EXPERTS];   // unbiased sigmoid scores
    __shared__ float s_masked[NT * E_EXPERTS];   // biased + group-masked (mutated)
    __shared__ float s_bias[E_EXPERTS];
    __shared__ float s_gs[NT * NGROUPS];
    __shared__ int   s_gmask[NT];
    __shared__ float s_redv[NT * NGROUPS];
    __shared__ int   s_redi[NT * NGROUPS];
    __shared__ int   s_sel[NT * TOPK];

    const int tid  = threadIdx.x;
    const int wave = tid >> 5;
    const int lane = tid & 31;
    const int tok0 = blockIdx.x * NT;

    s_bias[tid] = bias[tid];   // blockDim.x == 256 == E_EXPERTS

    // ------------- GEMM phase: logits[32 x 256], 2x2 tiles per wave -------------
    const int  mrow = lane & 15;
    const int  hf   = lane >> 4;
    const long arow0 = (long)(tok0 + mrow) * D_DIM;        // M tile 0
    const long arow1 = (long)(tok0 + 16 + mrow) * D_DIM;   // M tile 1
    const int  e0   = wave * 32;                           // experts [e0, e0+32)
    const long brow0 = (long)(e0 + mrow) * D_DIM;          // N tile 0
    const long brow1 = (long)(e0 + 16 + mrow) * D_DIM;     // N tile 1

    const float* pa0 = x + arow0 + hf * 8;
    const float* pa1 = x + arow1 + hf * 8;
    const float* pb0 = W + brow0 + hf * 8;
    const float* pb1 = W + brow1 + hf * 8;

    v8f c00 = {}; v8f c01 = {};
    v8f c10 = {}; v8f c11 = {};

#pragma unroll 2
    for (int k0 = 0; k0 < D_DIM; k0 += 32) {
        v16h a0 = load_a(pa0 + k0);
        v16h a1 = load_a(pa1 + k0);
        v16h b0 = load_b(pb0 + k0);
        v16h b1 = load_b(pb1 + k0);

        c00 = __builtin_amdgcn_wmma_f32_16x16x32_f16(false, a0, false, b0,
                                                     (short)0, c00, false, false);
        c01 = __builtin_amdgcn_wmma_f32_16x16x32_f16(false, a0, false, b1,
                                                     (short)0, c01, false, false);
        c10 = __builtin_amdgcn_wmma_f32_16x16x32_f16(false, a1, false, b0,
                                                     (short)0, c10, false, false);
        c11 = __builtin_amdgcn_wmma_f32_16x16x32_f16(false, a1, false, b1,
                                                     (short)0, c11, false, false);
    }

    // sigmoid -> LDS.  C/D layout: VGPR r, lanes 0-15: (M=r, N=lane);
    // lanes 16-31: (M=8+r, N=lane-16)
#pragma unroll
    for (int r = 0; r < 8; ++r) {
        int row0 = hf * 8 + r;          // within M tile 0
        int row1 = 16 + hf * 8 + r;     // within M tile 1
        s_scores[row0 * E_EXPERTS + e0 + mrow]      = sigmoidf(c00[r]);
        s_scores[row0 * E_EXPERTS + e0 + 16 + mrow] = sigmoidf(c01[r]);
        s_scores[row1 * E_EXPERTS + e0 + mrow]      = sigmoidf(c10[r]);
        s_scores[row1 * E_EXPERTS + e0 + 16 + mrow] = sigmoidf(c11[r]);
    }
    __syncthreads();

    // ---------------- Routing phase (32 tokens) ----------------
    // (A) per-(token,group) top-2 sum of biased scores -> group score
    //     32 tokens x 8 groups = 256 -> all threads active
    {
        int tm = tid >> 3, g = tid & 7;
        const float* sr = s_scores + tm * E_EXPERTS + g * GSIZE;
        const float* br = s_bias + g * GSIZE;
        float t1 = NEG_BIG, t2 = NEG_BIG;
        for (int i = 0; i < GSIZE; ++i) {
            float v = sr[i] + br[i];
            if (v > t1) { t2 = t1; t1 = v; }
            else if (v > t2) { t2 = v; }
        }
        s_gs[tm * NGROUPS + g] = t1 + t2;
    }
    __syncthreads();

    // (B) per-token top-4 groups -> bitmask (strict > keeps lowest index on ties)
    if (tid < NT) {
        float gv[NGROUPS];
        for (int g = 0; g < NGROUPS; ++g) gv[g] = s_gs[tid * NGROUPS + g];
        int gmask = 0;
        for (int k = 0; k < NLIM; ++k) {
            int bi = 0; float bv = gv[0];
            for (int g = 1; g < NGROUPS; ++g)
                if (gv[g] > bv) { bv = gv[g]; bi = g; }
            gmask |= (1 << bi);
            gv[bi] = NEG_BIG;
        }
        s_gmask[tid] = gmask;
    }
    __syncthreads();

    // (C) fill masked biased scores (one expert column per thread)
    {
        int e = tid;
        int g = e >> 5;
        float be = s_bias[e];
        for (int tm = 0; tm < NT; ++tm) {
            bool allowed = (s_gmask[tm] >> g) & 1;
            s_masked[tm * E_EXPERTS + e] =
                allowed ? (s_scores[tm * E_EXPERTS + e] + be) : NEG_BIG;
        }
    }
    __syncthreads();

    // (D) top-8 experts: 8 rounds of two-level argmax
    //     group level: 256 threads; merge: 32 threads
    for (int k = 0; k < TOPK; ++k) {
        {
            int tm = tid >> 3, g = tid & 7;
            const float* mr = s_masked + tm * E_EXPERTS + g * GSIZE;
            float bv = NEG_BIG; int bi = g * GSIZE;
            for (int i = 0; i < GSIZE; ++i)
                if (mr[i] > bv) { bv = mr[i]; bi = g * GSIZE + i; }
            s_redv[tm * NGROUPS + g] = bv;
            s_redi[tm * NGROUPS + g] = bi;
        }
        __syncthreads();
        if (tid < NT) {
            float bv = s_redv[tid * NGROUPS]; int bi = s_redi[tid * NGROUPS];
            for (int g = 1; g < NGROUPS; ++g) {
                float v = s_redv[tid * NGROUPS + g];
                if (v > bv) { bv = v; bi = s_redi[tid * NGROUPS + g]; }
            }
            s_sel[tid * TOPK + k] = bi;
            s_masked[tid * E_EXPERTS + bi] = NEG_BIG;
        }
        __syncthreads();
    }

    // (E) normalize (unbiased scores), write outputs, histogram
    if (tid < NT) {
        int tm = tid;
        int   si[TOPK];
        float sv[TOPK];
        float ssum = 1e-20f;
        for (int k = 0; k < TOPK; ++k) {
            si[k] = s_sel[tm * TOPK + k];
            sv[k] = s_scores[tm * E_EXPERTS + si[k]];
            ssum += sv[k];
        }
        float inv = RSCALE / ssum;
        long obase = (long)(tok0 + tm) * TOPK;
        for (int k = 0; k < TOPK; ++k) {
            out_scores[obase + k] = sv[k] * inv;
            out_sel[obase + k]    = (float)si[k];
            atomicAdd(&hist[si[k]], 1.0f);
        }
    }
}

extern "C" void kernel_launch(void* const* d_in, const int* in_sizes, int n_in,
                              void* d_out, int out_size, void* d_ws, size_t ws_size,
                              hipStream_t stream) {
    const float* x    = (const float*)d_in[0];   // (T, 2048) fp32
    const float* W    = (const float*)d_in[1];   // (256, 2048) fp32
    const float* bias = (const float*)d_in[2];   // (256,) fp32

    const int tokens = in_sizes[0] / D_DIM;      // 16384
    float* out        = (float*)d_out;
    float* out_scores = out;                              // T*8
    float* out_sel    = out + (long)tokens * TOPK;        // T*8
    float* hist       = out + 2L * tokens * TOPK;         // 256

    zero_hist_kernel<<<1, 256, 0, stream>>>(hist);
    router_fused_kernel<<<tokens / NT, 256, 0, stream>>>(
        x, W, bias, out_scores, out_sel, hist);
}
